// LayerFused_63711544869061
// MI455X (gfx1250) — matrix operand order repaired and checked by reading, not compiled, not run on hardware
//
#include <hip/hip_runtime.h>
#include <math.h>

// Problem dims (fixed by the reference)
#define NN 512
#define II 128
#define OO 128
#define PP 10

#define SQRT_2PI_F      2.5066282746310002f
#define INV_SQRT_2PI_F  0.3989422804014327f

// Per-pair table: 12 rows x 12 halves. Rows 0..9 = G, row 10 = w, row 11 = 0.
#define GW_STRIDE 144

typedef __attribute__((ext_vector_type(16))) _Float16 v16h;
typedef __attribute__((ext_vector_type(8)))  float    v8f;
typedef __attribute__((ext_vector_type(4)))  unsigned v4u;
typedef __attribute__((ext_vector_type(8)))  int      v8i;
typedef __attribute__((ext_vector_type(4)))  int      v4i;

union AFrag { v16h v; _Float16 h[16]; };
union CFrag { v8f  v; float    f[8];  };

// ---------------------------------------------------------------------------
// Kernel 1: per (i,o) pair — tanh(z), Q_hh + noise, Cholesky, L^-1,
// G = L^-T L^-1, w = G h.  One thread per pair (16384 threads; negligible).
// ws per pair: Gwh = 144 halves (12 rows x 12: rows 0..9 = G, 10 = w, 11 = 0)
//              scal = 16 floats  {z[0..9], l2, s2*|l|, sqrt(2pi)*s2*|l|}
// ---------------------------------------------------------------------------
__global__ void lf_prep_kernel(const float* __restrict__ z_raw,
                               const float* __restrict__ hmat,
                               const float* __restrict__ lvec,
                               const float* __restrict__ svec,
                               const float* __restrict__ jvec,
                               _Float16* __restrict__ Gwh,
                               float* __restrict__ scal) {
    int pair = blockIdx.x * blockDim.x + threadIdx.x;
    if (pair >= II * OO) return;

    float z[PP];
    for (int p = 0; p < PP; ++p) z[p] = tanhf(z_raw[(size_t)pair * PP + p]);

    float l   = lvec[pair];
    float l2  = l * l;
    float al  = fabsf(l);
    float s   = svec[pair];
    float s2  = s * s;
    float jit = jvec[pair];
    float invn = rsqrtf(l2) * INV_SQRT_2PI_F;

    float Q[PP][PP];
    for (int a = 0; a < PP; ++a)
        for (int b = 0; b <= a; ++b) {
            float d = z[a] - z[b];
            float val = invn * expf(-0.5f * d * d / l2);
            Q[a][b] = val; Q[b][a] = val;
        }
    float noise = (jit * jit + 0.001f) / (s2 * al * SQRT_2PI_F) + 0.001f;
    for (int a = 0; a < PP; ++a) Q[a][a] += noise;

    // Cholesky (lower, in place)
    for (int k = 0; k < PP; ++k) {
        float dk = sqrtf(Q[k][k]);
        Q[k][k] = dk;
        float inv = 1.0f / dk;
        for (int r = k + 1; r < PP; ++r) Q[r][k] *= inv;
        for (int r = k + 1; r < PP; ++r)
            for (int c = k + 1; c <= r; ++c)
                Q[r][c] -= Q[r][k] * Q[c][k];
    }

    // Invert lower-triangular L
    float Li[PP][PP];
    for (int c = 0; c < PP; ++c) {
        Li[c][c] = 1.0f / Q[c][c];
        for (int r = c + 1; r < PP; ++r) {
            float sum = 0.0f;
            for (int j = c; j < r; ++j) sum += Q[r][j] * Li[j][c];
            Li[r][c] = -sum / Q[r][r];
        }
    }

    // G = Li^T Li (symmetric)
    float G[PP][PP];
    for (int a = 0; a < PP; ++a)
        for (int b = 0; b <= a; ++b) {
            float sum = 0.0f;
            for (int r = a; r < PP; ++r) sum += Li[r][a] * Li[r][b];
            G[a][b] = sum; G[b][a] = sum;
        }

    // w = G h
    float hr[PP];
    for (int b = 0; b < PP; ++b) hr[b] = hmat[(size_t)pair * PP + b];
    float w[PP];
    for (int a = 0; a < PP; ++a) {
        float sum = 0.0f;
        for (int b = 0; b < PP; ++b) sum += G[a][b] * hr[b];
        w[a] = sum;
    }

    _Float16* g = Gwh + (size_t)pair * GW_STRIDE;
    for (int j = 0; j < GW_STRIDE; ++j) g[j] = (_Float16)0.0f;
    for (int a = 0; a < PP; ++a)
        for (int b = 0; b < PP; ++b) g[a * 12 + b] = (_Float16)G[a][b];
    for (int b = 0; b < PP; ++b) g[10 * 12 + b] = (_Float16)w[b];
    // row 11 stays zero: zero-source row for invalid lanes.

    float* sc = scal + (size_t)pair * 16;
    for (int p = 0; p < PP; ++p) sc[p] = z[p];
    sc[10] = l2;
    sc[11] = s2 * al;
    sc[12] = SQRT_2PI_F * s2 * al;
    sc[13] = 0.f; sc[14] = 0.f; sc[15] = 0.f;
}

// ---------------------------------------------------------------------------
// TDM helper: 2-D tensor_load_to_lds, tile = rows x width elements,
// row stride in elements, data_size_code: 1 = 2B, 2 = 4B.
// clang-22 (ROCm 7.2): 5-arg builtin; clang-23 (therock): 6-arg builtin.
// ---------------------------------------------------------------------------
#if __has_builtin(__builtin_amdgcn_tensor_load_to_lds)
#define LF_HAVE_TDM 1
__device__ __forceinline__ void tdm_load_2d(const void* gptr, unsigned lds_off,
                                            unsigned width, unsigned rows,
                                            unsigned row_stride,
                                            unsigned dsz_code) {
    unsigned long long ga = (unsigned long long)(uintptr_t)gptr;
    v4u g0;
    g0[0] = 1u;                                   // count=1, user descriptor
    g0[1] = lds_off;                              // LDS byte address
    g0[2] = (unsigned)ga;                         // global_addr[31:0]
    g0[3] = (unsigned)((ga >> 32) & 0x01ffffffu)  // global_addr[56:32]
          | (2u << 30);                           // type = 2 ("image")
    v8i g1;
    g1[0] = (int)(dsz_code << 16);                           // data_size
    g1[1] = (int)((width & 0xffffu) << 16);                  // tensor_dim0 lo
    g1[2] = (int)(((width >> 16) & 0xffffu) | ((rows & 0xffffu) << 16)); // td0 hi | td1 lo
    g1[3] = (int)(((rows >> 16) & 0xffffu) | ((width & 0xffffu) << 16)); // td1 hi | tile_dim0
    g1[4] = (int)(rows & 0xffffu);                           // tile_dim1 (tile_dim2=0)
    g1[5] = (int)row_stride;                                 // tensor_dim0_stride lo
    g1[6] = 0;
    g1[7] = 0;
    v4i gz4 = {0, 0, 0, 0};
#if defined(__clang_major__) && (__clang_major__ >= 23)
    v8i gz8 = {0, 0, 0, 0, 0, 0, 0, 0};
    __builtin_amdgcn_tensor_load_to_lds(g0, g1, gz4, gz4, gz8, 0);
#else
    __builtin_amdgcn_tensor_load_to_lds(g0, g1, gz4, gz4, 0);
#endif
}
#else
#define LF_HAVE_TDM 0
#endif

// ---------------------------------------------------------------------------
// Kernel 2: 4 waves/block share one `o`; each wave owns a 16-sample tile.
// Per-o tables staged into LDS via the Tensor Data Mover, then per i:
//   WMMA#1: U  = Q * [G | w | 0]   (16x16x32 f16 -> f32)
//   WMMA#2: Qc = Q * I             (Q re-laid into C-layout)
//   t5[m] += c_i * sum_{c<10} U[m,c]*Qc[m,c];  mean[m] += U[m,10]
// ---------------------------------------------------------------------------
__global__ __launch_bounds__(128) void lf_main_kernel(
        const float*    __restrict__ x_mean, const float* __restrict__ x_var,
        const _Float16* __restrict__ Gwh,    const float* __restrict__ scal,
        float* __restrict__ out_mean,        float* __restrict__ out_var) {

    __shared__ _Float16 s_gwh[II * GW_STRIDE];   // 36864 B
    __shared__ float    s_sc [II * 16];          //  8192 B

    const int  o     = blockIdx.x & (OO - 1);
    const int  tg    = blockIdx.x >> 7;        // 0..7
    const int  wave  = threadIdx.x >> 5;
    const int  lane  = threadIdx.x & 31;
    const int  tile  = tg * 4 + wave;          // 0..31
    const int  n0    = tile * 16;
    const int  m     = lane & 15;
    const bool upper = lane >= 16;

    // ---- stage per-o tables into LDS (TDM; cooperative-copy fallback) ----
#if LF_HAVE_TDM
    if (threadIdx.x < 32) {
        tdm_load_2d(Gwh + (size_t)o * GW_STRIDE, (unsigned)(uintptr_t)&s_gwh[0],
                    /*width*/GW_STRIDE, /*rows*/II, /*stride*/OO * GW_STRIDE, /*2B*/1);
        tdm_load_2d(scal + (size_t)o * 16, (unsigned)(uintptr_t)&s_sc[0],
                    /*width*/16,  /*rows*/II, /*stride*/OO * 16,  /*4B*/2);
        __builtin_amdgcn_s_wait_tensorcnt(0);
    }
#else
    for (int idx = threadIdx.x; idx < II * GW_STRIDE; idx += 128) {
        int i = idx / GW_STRIDE, j = idx - i * GW_STRIDE;
        s_gwh[idx] = Gwh[((size_t)i * OO + o) * GW_STRIDE + j];
    }
    for (int idx = threadIdx.x; idx < II * 16; idx += 128) {
        int i = idx >> 4, j = idx & 15;
        s_sc[idx] = scal[((size_t)i * OO + o) * 16 + j];
    }
#endif
    __syncthreads();

    const float* xm_row = x_mean + (size_t)(n0 + m) * II;
    const float* xv_row = x_var  + (size_t)(n0 + m) * II;
    __builtin_prefetch(xm_row, 0, 0);
    __builtin_prefetch(xv_row, 0, 0);

    // Identity B fragment: B[k][c] = (k==c); half h element j (<8) -> row 8h+j, col m.
    AFrag bid;
    #pragma unroll
    for (int j = 0; j < 16; ++j) bid.h[j] = (_Float16)0.0f;
    if (upper == (m >= 8)) bid.h[m & 7] = (_Float16)1.0f;

    const float    t3m  = upper ? 0.0f : 1.0f;
    // Invalid B lanes (column > 10) read zero-row 11.
    const int      brow = ((m <= PP) ? m : 11) * 12;
    const _Float16 hz   = (_Float16)0.0f;

    float acc_q[8], acc_mn[8];
    #pragma unroll
    for (int r = 0; r < 8; ++r) { acc_q[r] = 0.f; acc_mn[r] = 0.f; }
    float acc_t3 = 0.f;

    // A-layout: lane<16 halves j=0..7 -> k=j; lane>=16 halves j=0..7 -> k=8+j
    // (halves 8..15 map to k>=16 which is always zero-padded).
    const int kb = upper ? 8 : 0;

    for (int i = 0; i < II; ++i) {
        const float* sc = s_sc + i * 16;
        float l2 = sc[10];
        float sa = sc[11];                  // s^2 * |l|
        float ci = sc[12];                  // sqrt(2pi) * s^2 * |l|
        float xm = xm_row[i];
        float xv = xv_row[i];
        float v  = xv + l2;
        float inv  = __builtin_amdgcn_rsqf(v) * INV_SQRT_2PI_F;
        float nh   = -0.5f * __builtin_amdgcn_rcpf(v);
        // upper half only has valid k (=8,9) at j=0,1; mask the rest via inv2.
        float inv2 = upper ? 0.0f : inv;

        // A fragment: branchless, 8 exps per lane.
        AFrag a;
        #pragma unroll
        for (int j = 0; j < 8; ++j) {
            float d  = xm - sc[j + kb];            // j+kb <= 15, in bounds
            float sj = (j < 2) ? inv : inv2;
            a.h[j] = (_Float16)(sj * __expf(nh * d * d));
        }
        #pragma unroll
        for (int j = 8; j < 16; ++j) a.h[j] = hz;

        // B fragment from LDS: 12 unconditional half loads (row 11 = zeros).
        AFrag b;
        const _Float16* grow = s_gwh + i * GW_STRIDE + brow;
        #pragma unroll
        for (int j = 0; j < 8; ++j) b.h[j] = (kb + j < 12) ? grow[kb + j] : hz;
        #pragma unroll
        for (int j = 8; j < 16; ++j) b.h[j] = hz;

        v8f zero8 = {};
        CFrag u, qc;
        u.v  = __builtin_amdgcn_wmma_f32_16x16x32_f16(
                   false, a.v, false, b.v,   (short)0, zero8, false, false);
        qc.v = __builtin_amdgcn_wmma_f32_16x16x32_f16(
                   false, a.v, false, bid.v, (short)0, zero8, false, false);

        // lane holds column m of U/Qc, rows r (+8 for the upper half).
        float selq = (m < PP)  ? ci  : 0.f;
        float selm = (m == PP) ? 1.f : 0.f;
        #pragma unroll
        for (int r = 0; r < 8; ++r) {
            acc_q[r]  = fmaf(selq * u.f[r], qc.f[r], acc_q[r]);
            acc_mn[r] = fmaf(selm, u.f[r], acc_mn[r]);
        }
        acc_t3 += t3m * sa * __builtin_amdgcn_rsqf(l2 + 2.0f * xv);
    }

    // Reduce across the 16 lanes of each half-wave (wave32: xor 1,2,4,8).
    #pragma unroll
    for (int off = 1; off < 16; off <<= 1) {
        #pragma unroll
        for (int r = 0; r < 8; ++r) {
            acc_q[r]  += __shfl_xor(acc_q[r],  off, 32);
            acc_mn[r] += __shfl_xor(acc_mn[r], off, 32);
        }
    }
    // Pull upper-half rows (8..15) into the lower half.
    float oq[8], om[8];
    #pragma unroll
    for (int r = 0; r < 8; ++r) {
        oq[r] = __shfl_xor(acc_q[r],  16, 32);
        om[r] = __shfl_xor(acc_mn[r], 16, 32);
    }

    if (!upper) {
        int row = m;                         // lane m writes sample row m
        float t5c = 0.f, mn = 0.f;
        #pragma unroll
        for (int r = 0; r < 8; ++r) {
            t5c = (row == r)     ? acc_q[r]  : t5c;
            mn  = (row == r)     ? acc_mn[r] : mn;
            t5c = (row == r + 8) ? oq[r]     : t5c;
            mn  = (row == r + 8) ? om[r]     : mn;
        }
        float var = acc_t3 - t5c + 0.1f * (float)II;
        out_mean[(size_t)(n0 + row) * OO + o] = mn;
        out_var [(size_t)(n0 + row) * OO + o] = var;
    }
}

// ---------------------------------------------------------------------------
extern "C" void kernel_launch(void* const* d_in, const int* in_sizes, int n_in,
                              void* d_out, int out_size, void* d_ws, size_t ws_size,
                              hipStream_t stream) {
    (void)in_sizes; (void)n_in; (void)out_size; (void)ws_size;
    const float* x_mean = (const float*)d_in[0];
    const float* x_var  = (const float*)d_in[1];
    const float* z_raw  = (const float*)d_in[2];
    const float* hmat   = (const float*)d_in[3];
    const float* lvec   = (const float*)d_in[4];
    const float* svec   = (const float*)d_in[5];
    const float* jvec   = (const float*)d_in[6];

    float* out_mean = (float*)d_out;
    float* out_var  = out_mean + (size_t)NN * OO;

    // Workspace: Gwh = 16384*144 halves (4.7 MB), scal = 16384*16 floats (1 MB)
    _Float16* Gwh = (_Float16*)d_ws;
    float* scal = (float*)((char*)d_ws + (size_t)II * OO * GW_STRIDE * sizeof(_Float16));

    lf_prep_kernel<<<(II * OO + 127) / 128, 128, 0, stream>>>(
        z_raw, hmat, lvec, svec, jvec, Gwh, scal);

    // 128 o * 8 tile-groups = 1024 blocks; 4 waves/block share one o.
    lf_main_kernel<<<1024, 128, 0, stream>>>(
        x_mean, x_var, Gwh, scal, out_mean, out_var);
}
